// MeanAttn_40381282517304
// MI455X (gfx1250) — matrix operand order repaired and checked by reading, not compiled, not run on hardware
//
#include <hip/hip_runtime.h>
#include <hip/hip_bf16.h>
#include <math.h>

typedef __attribute__((ext_vector_type(16))) _Float16 v16h;
typedef __attribute__((ext_vector_type(8)))  _Float16 v8h;
typedef __attribute__((ext_vector_type(8)))  float    v8f;

// ---------------------------------------------------------------- WMMA helpers
__device__ inline v8f wmma_f16f32(v16h a, v16h b, v8f c) {
  // v_wmma_f32_16x16x32_f16, f32 accumulate
  return __builtin_amdgcn_wmma_f32_16x16x32_f16(false, a, false, b, (short)0, c,
                                                false, false);
}

__device__ inline v8f v8f_zero() {
  v8f z;
#pragma unroll
  for (int e = 0; e < 8; ++e) z[e] = 0.0f;
  return z;
}

// A fragment (16x32 f16) per ISA layout, two 16B loads (LDS or global).
// lane<16: row=rbase+lane,    elems 0..7 -> K0..7,  8..15 -> K16..23
// lane>=16: row=rbase+lane-16, elems 0..7 -> K8..15, 8..15 -> K24..31
// Requires ld (halfs) multiple of 8 so 16B alignment holds.
__device__ inline v16h fragA16(const _Float16* base, int ld, int rbase, int kb0,
                               int lane) {
  int r = rbase + (lane & 15);
  int koff = kb0 + ((lane & 16) ? 8 : 0);
  const _Float16* p = base + (size_t)r * ld + koff;
  v8h a = *(const v8h*)(p);
  v8h b = *(const v8h*)(p + 16);
  v16h f;
#pragma unroll
  for (int e = 0; e < 8; ++e) { f[e] = a[e]; f[8 + e] = b[e]; }
  return f;
}

// B fragment (32x16 f16), unguarded: lane = column n (source rows pre-padded so
// every access is in-bounds); lane<16 elems 0..7->K0..7, 8..15->K16..23;
// lane>=16 elems 0..7->K8..15, 8..15->K24..31.
// Source f16 row-major (N,K): two runs of 8 contiguous K per lane (k0, k0+16) -> two 16B loads.
__device__ inline v16h fragB16u(const _Float16* src, int ld, int cb, int kb0,
                                int lane) {
  int n = cb + (lane & 15);
  int k0 = kb0 + ((lane & 16) ? 8 : 0);
  const _Float16* p = src + (size_t)n * ld + k0;
  v8h a = *(const v8h*)(p);
  v8h b = *(const v8h*)(p + 16);
  v16h f;
#pragma unroll
  for (int e = 0; e < 8; ++e) { f[e] = a[e]; f[8 + e] = b[e]; }
  return f;
}

// ------------------------------------------------------------- token <-> pixel
__device__ inline int tok_of(int b, int y, int x, int nWw, int nWh) {
  return (((b * nWh + (y >> 3)) * nWw + (x >> 3)) << 6) + ((y & 7) << 3) + (x & 7);
}
__device__ inline void tok_dec(int tk, int nWw, int nWh, int& b, int& y, int& x) {
  int w = tk >> 6, l = tk & 63;
  int ww = w % nWw;
  int t2 = w / nWw;
  int wh = t2 % nWh;
  b = t2 / nWh;
  y = (wh << 3) + (l >> 3);
  x = (ww << 3) + (l & 7);
}

// ---------------------------------------------------------------- GEMM kernel
// C[M,N] = A[M,K](f16) * W16[Npad,K]^T * scale + bias, 64x64 tiles, no LDS:
// 128 threads = 4 waves, wave w owns rows w*16..w*16+15 (disjoint -> direct
// global fragment loads, no staging barriers). W16 rows padded to >= 64-tile.
struct GemmP {
  const _Float16* A;
  const _Float16* W;   // f16 (Npad,K) row-major, zero-padded rows
  const float* bias;
  int M, N, K;
  float scale;
  int mode;        // 0: f16 out  1: f32 out  2: ffn-in split  3: nchw scatter
  _Float16* o16;   // mode0 (ld=N) / mode2 raw x1 (ld=96)
  float* o32;      // mode1 (ld=N) / mode3 NCHW base
  _Float16* o16b;  // mode2 gelu half (ld=192)
  int dh, dw, nWw, nWh;  // mode3 geometry
};

__global__ __launch_bounds__(128) void gemm_wmma(GemmP p) {
  const int tid = threadIdx.x;
  const int lane = tid & 31, wv = tid >> 5;
  const int n0 = blockIdx.y * 64;
  const int rbase = blockIdx.x * 64 + wv * 16;
  v8f acc[4];
#pragma unroll
  for (int i = 0; i < 4; ++i) acc[i] = v8f_zero();

  for (int kb = 0; kb < p.K; kb += 32) {
    v16h fa = fragA16(p.A, p.K, rbase, kb, lane);
#pragma unroll
    for (int nt = 0; nt < 4; ++nt) {
      v16h fb = fragB16u(p.W, p.K, n0 + nt * 16, kb, lane);
      acc[nt] = wmma_f16f32(fa, fb, acc[nt]);
    }
  }

  const int chi = (lane & 16) ? 8 : 0;
  const int cn = lane & 15;
#pragma unroll
  for (int nt = 0; nt < 4; ++nt) {
    int col = n0 + nt * 16 + cn;
    if (col >= p.N) continue;
    float bia = p.bias ? p.bias[col] : 0.0f;
#pragma unroll
    for (int e = 0; e < 8; ++e) {
      int row = rbase + chi + e;
      float v = acc[nt][e] * p.scale + bia;
      if (p.mode == 0) {
        p.o16[(size_t)row * p.N + col] = (_Float16)v;
      } else if (p.mode == 1) {
        p.o32[(size_t)row * p.N + col] = v;
      } else if (p.mode == 2) {
        if (col < 96) {
          p.o16[(size_t)row * 96 + col] = (_Float16)v;  // raw x1 for dwconv
        } else {
          float g = 0.5f * v * (1.0f + erff(v * 0.7071067811865475f));  // exact gelu
          p.o16b[(size_t)row * 192 + col] = (_Float16)g;
        }
      } else {  // window-reverse NCHW scatter
        int b, y, xx;
        tok_dec(row, p.nWw, p.nWh, b, y, xx);
        p.o32[(((size_t)b * 96 + col) * p.dh + y) * p.dw + xx] = v;
      }
    }
  }
}

// ------------------------------------------------- fused per-window attention
// one block (4 waves) per 8x8 window: logits = q*k^T*scl (LDS), row softmax,
// o = P*v + lepe, all WMMA; LDS only for the cross-wave softmax exchange.
__global__ __launch_bounds__(128) void attn_win_kernel(const _Float16* q16,
                                                       const _Float16* k16,
                                                       const _Float16* vT16,
                                                       const float* lepe32,
                                                       _Float16* o16, float scl) {
  __shared__ float lg[64 * 64];
  __shared__ __align__(16) _Float16 ap[64 * 64];
  const int tid = threadIdx.x;
  const int lane = tid & 31, wv = tid >> 5;
  const size_t wbase = (size_t)blockIdx.x * 64;

  const int rbase = wv * 16;
  const int chi = (lane & 16) ? 8 : 0;
  const int cn = lane & 15;

  v8f acc[4];
#pragma unroll
  for (int i = 0; i < 4; ++i) acc[i] = v8f_zero();
  const _Float16* qw = q16 + wbase * 96;  // this window's 64x96 q rows
#pragma unroll
  for (int kb = 0; kb < 96; kb += 32) {
    v16h fa = fragA16(qw, 96, rbase, kb, lane);
#pragma unroll
    for (int nt = 0; nt < 4; ++nt) {
      v16h fb = fragB16u(k16, 96, nt * 16, kb, lane);  // B = k^T
      acc[nt] = wmma_f16f32(fa, fb, acc[nt]);
    }
  }
#pragma unroll
  for (int nt = 0; nt < 4; ++nt)
#pragma unroll
    for (int e = 0; e < 8; ++e)
      lg[(rbase + chi + e) * 64 + nt * 16 + cn] = acc[nt][e] * scl;
  __syncthreads();

  if (tid < 64) {  // one thread per query row
    float mx = -1e30f;
    for (int t = 0; t < 64; ++t) mx = fmaxf(mx, lg[tid * 64 + t]);
    float s = 0.0f;
    for (int t = 0; t < 64; ++t) {
      float e = __expf(lg[tid * 64 + t] - mx);
      lg[tid * 64 + t] = e;
      s += e;
    }
    float inv = 1.0f / s;
    for (int t = 0; t < 64; ++t) ap[tid * 64 + t] = (_Float16)(lg[tid * 64 + t] * inv);
  }
  __syncthreads();

  v8f acc2[6];
#pragma unroll
  for (int i = 0; i < 6; ++i) acc2[i] = v8f_zero();
#pragma unroll
  for (int kb = 0; kb < 64; kb += 32) {
    v16h fa = fragA16(ap, 64, rbase, kb, lane);
#pragma unroll
    for (int nt = 0; nt < 6; ++nt) {
      // B[k=t][n=c] = vT16[c*64+t]: contiguous along K per lane
      v16h fb = fragB16u(vT16, 64, nt * 16, kb, lane);
      acc2[nt] = wmma_f16f32(fa, fb, acc2[nt]);
    }
  }
#pragma unroll
  for (int nt = 0; nt < 6; ++nt) {
    int col = nt * 16 + cn;
#pragma unroll
    for (int e = 0; e < 8; ++e) {
      int r = rbase + chi + e;
      o16[(wbase + r) * 96 + col] = (_Float16)(acc2[nt][e] + lepe32[r * 96 + col]);
    }
  }
}

// --------------------------------------------------------------- small kernels
__global__ void zero_kernel(float* p, int n) {
  int i = blockIdx.x * 256 + threadIdx.x;
  if (i < n) p[i] = 0.0f;
}

// f32 -> f16 with zero row padding: identity map for i < nValid, pad with 0.
__global__ void cvtpad_kernel(const float* in, _Float16* out, int nValid,
                              int nTot) {
  int i = blockIdx.x * 256 + threadIdx.x;
  if (i < nTot) out[i] = (i < nValid) ? (_Float16)in[i] : (_Float16)0.0f;
}

// v (64,96) f32 -> vT (96,64) f16
__global__ void vtrans_kernel(const float* v32, _Float16* vT16) {
  int i = blockIdx.x * 256 + threadIdx.x;  // 6144
  int t = i & 63, c = i >> 6;
  vT16[c * 64 + t] = (_Float16)v32[t * 96 + c];
}

// xn = normalize(x NCHW over C) -> f16 row-major (B*HW, 192); coalesced reads,
// 16B vector row writes.
__global__ void xn_norm_kernel(const float* x, _Float16* xn16) {
  const int HW = 16384, C = 192;
  int b = blockIdx.y;
  int pix = blockIdx.x * 256 + threadIdx.x;
  const float* xb = x + (size_t)b * C * HW + pix;
  float s = 0.0f;
  for (int c = 0; c < C; ++c) {
    float v = xb[(size_t)c * HW];
    s += v * v;
  }
  float rn = 1.0f / fmaxf(sqrtf(s), 1e-12f);
  _Float16* o = xn16 + (size_t)(b * HW + pix) * C;
  for (int c = 0; c < C; c += 8) {
    v8h chunk;
#pragma unroll
    for (int e = 0; e < 8; ++e)
      chunk[e] = (_Float16)(xb[(size_t)(c + e) * HW] * rn);
    *(v8h*)&o[c] = chunk;
  }
}

// wave-per-row L2 normalize of (64,192) f32 -> f32 + f16
__global__ void rownorm192_kernel(const float* in, float* out, _Float16* out16) {
  int row = blockIdx.x * 8 + (threadIdx.x >> 5);
  int lane = threadIdx.x & 31;
  float s = 0.0f;
  for (int c = lane; c < 192; c += 32) {
    float v = in[row * 192 + c];
    s += v * v;
  }
#pragma unroll
  for (int off = 16; off > 0; off >>= 1) s += __shfl_xor(s, off, 32);
  float rn = 1.0f / fmaxf(sqrtf(s), 1e-12f);
  for (int c = lane; c < 192; c += 32) {
    float v = in[row * 192 + c] * rn;
    out[row * 192 + c] = v;
    out16[row * 192 + c] = (_Float16)v;
  }
}

// per-pixel softmax over 64 means, top-3, prob-weighted segment sums.
// LDS-local 64x192 accumulators, one global flush per block.
__global__ void assign_kernel(const float* sim, const _Float16* xn16, float* newm,
                              float* denom) {
  __shared__ float accs[64 * 192];
  __shared__ float dens[64];
  const int tid = threadIdx.x;
  for (int i = tid; i < 64 * 192; i += 256) accs[i] = 0.0f;
  if (tid < 64) dens[tid] = 0.0f;
  __syncthreads();

  const size_t p = (size_t)blockIdx.x * 256 + tid;
  const float* sp = sim + p * 64;
  float v1 = -1e30f, v2 = -1e30f, v3 = -1e30f;
  int i1 = 0, i2 = 0, i3 = 0;
  for (int t = 0; t < 64; ++t) {
    float v = sp[t];
    if (v > v1) { v3 = v2; i3 = i2; v2 = v1; i2 = i1; v1 = v; i1 = t; }
    else if (v > v2) { v3 = v2; i3 = i2; v2 = v; i2 = t; }
    else if (v > v3) { v3 = v; i3 = t; }
  }
  float s = 0.0f;
  for (int t = 0; t < 64; ++t) s += __expf(sp[t] - v1);
  float inv = 1.0f / s;
  float p1 = inv;                    // exp(v1-v1)=1
  float p2 = __expf(v2 - v1) * inv;
  float p3 = __expf(v3 - v1) * inv;

  const _Float16* xr = xn16 + p * 192;
  for (int c = 0; c < 192; ++c) {
    float xv = (float)xr[c];
    atomicAdd(&accs[i1 * 192 + c], p1 * xv);
    atomicAdd(&accs[i2 * 192 + c], p2 * xv);
    atomicAdd(&accs[i3 * 192 + c], p3 * xv);
  }
  atomicAdd(&dens[i1], p1);
  atomicAdd(&dens[i2], p2);
  atomicAdd(&dens[i3], p3);
  __syncthreads();

  for (int i = tid; i < 64 * 192; i += 256) atomicAdd(&newm[i], accs[i]);
  if (tid < 64) atomicAdd(&denom[tid], dens[tid]);
}

// m = normalize(where(denom>1e-6, newm/(denom+1e-6), mn)); wave per row
__global__ void finalize_means_kernel(const float* newm, const float* denom,
                                      const float* mn, float* mout,
                                      _Float16* mout16) {
  int row = blockIdx.x * 8 + (threadIdx.x >> 5);
  int lane = threadIdx.x & 31;
  float d = denom[row];
  bool ok = d > 1e-6f;
  float rd = 1.0f / (d + 1e-6f);
  float s = 0.0f;
  for (int c = lane; c < 192; c += 32) {
    float v = ok ? newm[row * 192 + c] * rd : mn[row * 192 + c];
    s += v * v;
  }
#pragma unroll
  for (int off = 16; off > 0; off >>= 1) s += __shfl_xor(s, off, 32);
  float rn = 1.0f / fmaxf(sqrtf(s), 1e-12f);
  for (int c = lane; c < 192; c += 32) {
    float v = (ok ? newm[row * 192 + c] * rd : mn[row * 192 + c]) * rn;
    mout[row * 192 + c] = v;
    mout16[row * 192 + c] = (_Float16)v;
  }
}

__global__ void momentum_kernel(const float* running, const float* m, float* rm) {
  int i = blockIdx.x * 256 + threadIdx.x;
  if (i < 64 * 192) rm[i] = 0.99f * running[i] + 0.01f * m[i];
}

__global__ void slice_mi_kernel(const float* rm, _Float16* mi16, int off) {
  int i = blockIdx.x * 256 + threadIdx.x;  // 64*96
  int c = i % 96, t = i / 96;
  mi16[i] = (_Float16)rm[t * 192 + off + c];
}

// dsx NCHW f32 -> window-token-ordered f16 rows (tokens, 96)
__global__ void winpart_kernel(const float* dsx, _Float16* xw16, int dh, int dw,
                               int nWw, int nWh) {
  int idx = blockIdx.x * 256 + threadIdx.x;
  int c = idx % 96, tk = idx / 96;
  int b, y, x;
  tok_dec(tk, nWw, nWh, b, y, x);
  xw16[idx] = (_Float16)dsx[(((size_t)b * 96 + c) * dh + y) * dw + x];
}

// lepe = dwconv3x3(v reshaped (96,8,8)) + bias ; v,lepe are (64,96) f32
__global__ void lepe3_kernel(const float* v32, const float* kern, const float* bias,
                             float* lepe32) {
  int idx = blockIdx.x * 256 + threadIdx.x;  // 64*96
  int c = idx % 96, t = idx / 96;
  int y = t >> 3, x = t & 7;
  float s = 0.0f;
#pragma unroll
  for (int dy = 0; dy < 3; ++dy)
#pragma unroll
    for (int dx = 0; dx < 3; ++dx) {
      int ny = y + dy - 1, nx = x + dx - 1;
      if (ny >= 0 && ny < 8 && nx >= 0 && nx < 8)
        s += kern[c * 9 + dy * 3 + dx] * v32[(ny * 8 + nx) * 96 + c];
    }
  lepe32[t * 96 + c] = s + bias[c];
}

// channel LayerNorm over 96, wave per row (96 = 3 x 32 lanes)
__global__ void ln_kernel(const float* in, const float* lw, const float* lb,
                          _Float16* out) {
  int row = blockIdx.x * 8 + (threadIdx.x >> 5);
  int lane = threadIdx.x & 31;
  const float* r = in + (size_t)row * 96;
  float x0 = r[lane], x1 = r[lane + 32], x2 = r[lane + 64];
  float s = x0 + x1 + x2;
#pragma unroll
  for (int off = 16; off > 0; off >>= 1) s += __shfl_xor(s, off, 32);
  float u = s * (1.0f / 96.0f);
  float d0 = x0 - u, d1 = x1 - u, d2 = x2 - u;
  float q = d0 * d0 + d1 * d1 + d2 * d2;
#pragma unroll
  for (int off = 16; off > 0; off >>= 1) q += __shfl_xor(q, off, 32);
  float rstd = rsqrtf(q * (1.0f / 96.0f) + 1e-6f);
  _Float16* o = out + (size_t)row * 96;
  o[lane]      = (_Float16)(lw[lane]      * d0 * rstd + lb[lane]);
  o[lane + 32] = (_Float16)(lw[lane + 32] * d1 * rstd + lb[lane + 32]);
  o[lane + 64] = (_Float16)(lw[lane + 64] * d2 * rstd + lb[lane + 64]);
}

// x1 = dwconv5x5(h1) + bias, written into cols 0..95 of hcat (ld 192)
__global__ void dwconv5_kernel(const _Float16* h1, const float* kern,
                               const float* bias, _Float16* hcat, int dh, int dw,
                               int nWw, int nWh) {
  int idx = blockIdx.x * 256 + threadIdx.x;
  int c = idx % 96, tk = idx / 96;
  int b, y, x;
  tok_dec(tk, nWw, nWh, b, y, x);
  float s = 0.0f;
#pragma unroll
  for (int dy = 0; dy < 5; ++dy) {
    int ny = y + dy - 2;
    if (ny < 0 || ny >= dh) continue;
#pragma unroll
    for (int dx = 0; dx < 5; ++dx) {
      int nx = x + dx - 2;
      if (nx < 0 || nx >= dw) continue;
      int nt = tok_of(b, ny, nx, nWw, nWh);
      s += kern[c * 25 + dy * 5 + dx] * (float)h1[(size_t)nt * 96 + c];
    }
  }
  hcat[(size_t)tk * 192 + c] = (_Float16)(s + bias[c]);
}

// ------------------------------------------------------------------ host side
static void launch_gemm(hipStream_t s, const _Float16* A, const _Float16* W,
                        const float* bias, int M, int N, int K, float scale,
                        int mode, _Float16* o16, float* o32, _Float16* o16b,
                        int dh, int dw, int nWw, int nWh) {
  GemmP p;
  p.A = A; p.W = W; p.bias = bias; p.M = M; p.N = N; p.K = K; p.scale = scale;
  p.mode = mode; p.o16 = o16; p.o32 = o32; p.o16b = o16b;
  p.dh = dh; p.dw = dw; p.nWw = nWw; p.nWh = nWh;
  dim3 g(M / 64, (N + 63) / 64);
  gemm_wmma<<<g, 128, 0, s>>>(p);
}

extern "C" void kernel_launch(void* const* d_in, const int* in_sizes, int n_in,
                              void* d_out, int out_size, void* d_ws, size_t ws_size,
                              hipStream_t stream) {
  (void)n_in; (void)out_size; (void)ws_size;
  const float* x       = (const float*)d_in[0];
  const float* dsx0    = (const float*)d_in[1];
  const float* dsx1    = (const float*)d_in[2];
  const float* means   = (const float*)d_in[3];
  const float* running = (const float*)d_in[4];

  // Param flattening order: jax-sorted dict keys vs insertion order (auto-detect).
  // logical: 0 qw 1 qb 2 kw 3 kb 4 vw 5 vb 6 pw 7 pb 8 lepe_w 9 lepe_b
  //          10 ln_w 11 ln_b 12 in_w 13 in_b 14 dwc_w 15 dwc_b 16 out_w 17 out_b
  static const int mapSorted[18] = {15, 14, 5, 4, 17, 16, 13, 12, 7, 6,
                                    9,  8,  3, 2, 1,  0,  11, 10};
  bool sortedOrder = (in_sizes[5] == 96 && in_sizes[6] == 2400);
  const float* prm[2][18];
  for (int br = 0; br < 2; ++br)
    for (int j = 0; j < 18; ++j)
      prm[br][j] = (const float*)d_in[5 + 18 * br + (sortedOrder ? mapSorted[j] : j)];

  // ---- workspace layout (phase-aliased) ----
  char* w = (char*)d_ws;
  const size_t SZ_A = 50331648;   // 131072*192*2
  const size_t SZ_B = 33554432;   // 131072*64*4
  const size_t SZ_C = 50331648;   // 131072*96*4 == 131072*192*2
  _Float16* xn16   = (_Float16*)(w);                       // phase A
  float*    sim32  = (float*)(w + SZ_A);                   // phase A
  _Float16* xw16   = (_Float16*)(w);                       // phase B (xn dead)
  _Float16* o16buf = (_Float16*)(w + SZ_A / 2);
  _Float16* ln16   = (_Float16*)(w);                       // xw dead after q
  _Float16* h1_16  = (_Float16*)(w + SZ_A / 2);            // o dead after proj
  _Float16* q16    = (_Float16*)(w + SZ_A);                // sim dead
  float*    proj32 = (float*)(w + SZ_A + SZ_B);            // region C
  _Float16* hcat16 = (_Float16*)(w + SZ_A + SZ_B);         // proj dead after LN

  // small-buffer region D (bump allocated, 256B aligned)
  char* D = w + SZ_A + SZ_B + SZ_C;
  size_t doff = 0;
  auto dalloc = [&](size_t bytes) -> char* {
    char* p = D + doff;
    doff += (bytes + 255) & ~(size_t)255;
    return p;
  };
  float*    mnA    = (float*)dalloc(49152);
  float*    mnB    = (float*)dalloc(49152);
  _Float16* mnA16  = (_Float16*)dalloc(24576);
  _Float16* mnB16  = (_Float16*)dalloc(24576);
  float*    newm   = (float*)dalloc(49152);
  float*    denom  = (float*)dalloc(256);
  _Float16* mi16   = (_Float16*)dalloc(12288);
  _Float16* k16    = (_Float16*)dalloc(12288);
  float*    v32    = (float*)dalloc(24576);
  _Float16* vT16   = (_Float16*)dalloc(12288);
  float*    lepe32 = (float*)dalloc(24576);
  _Float16* wq16   = (_Float16*)dalloc(24576);   // 128x96 (zero-padded)
  _Float16* wk16   = (_Float16*)dalloc(24576);   // 128x96
  _Float16* wv16   = (_Float16*)dalloc(24576);   // 128x96
  _Float16* wp16   = (_Float16*)dalloc(24576);   // 128x96
  _Float16* win16  = (_Float16*)dalloc(36864);   // 192x96 (no pad needed)
  _Float16* wot16  = (_Float16*)dalloc(49152);   // 128x192 (zero-padded)

  float* out = (float*)d_out;
  float* out_b0 = out;                    // (8,96,128,128)
  float* out_b1 = out + 12582912;         // (8,96,64,64)
  float* rm     = out + 15728640;         // (64,192)

  const int NPIX = 131072;  // 8 * 128 * 128

  // ======== Phase A: soft k-means over normalized pixels ========
  xn_norm_kernel<<<dim3(64, 8), 256, 0, stream>>>(x, xn16);
  rownorm192_kernel<<<8, 256, 0, stream>>>(means, mnA, mnA16);
  float* mcur = mnA;     _Float16* mcur16 = mnA16;
  float* mnext = mnB;    _Float16* mnext16 = mnB16;
  const float simScale = sqrtf(192.0f);
  for (int it = 0; it < 4; ++it) {
    zero_kernel<<<(12352 + 255) / 256, 256, 0, stream>>>(newm, 12352);
    launch_gemm(stream, xn16, mcur16, nullptr, NPIX, 64, 192, simScale, 1,
                nullptr, sim32, nullptr, 0, 0, 0, 0);
    assign_kernel<<<NPIX / 256, 256, 0, stream>>>(sim32, xn16, newm, denom);
    finalize_means_kernel<<<8, 256, 0, stream>>>(newm, denom, mcur, mnext,
                                                 mnext16);
    float* t = mcur; mcur = mnext; mnext = t;
    _Float16* t16 = mcur16; mcur16 = mnext16; mnext16 = t16;
  }
  momentum_kernel<<<48, 256, 0, stream>>>(running, mcur, rm);

  // ======== Phase B: two attention + FFN branches ========
  for (int br = 0; br < 2; ++br) {
    const int dh = br ? 64 : 128, dw = dh;
    const int nWh = dh / 8, nWw = dw / 8;
    const int M = 8 * dh * dw;
    const float* dsx = br ? dsx1 : dsx0;
    float* outb = br ? out_b1 : out_b0;
    const float* qw = prm[br][0],  *qb = prm[br][1];
    const float* kw = prm[br][2],  *kb = prm[br][3];
    const float* vw = prm[br][4],  *vb = prm[br][5];
    const float* pw = prm[br][6],  *pb = prm[br][7];
    const float* lpw = prm[br][8], *lpb = prm[br][9];
    const float* lnw = prm[br][10], *lnb = prm[br][11];
    const float* inw = prm[br][12], *inb = prm[br][13];
    const float* dcw = prm[br][14], *dcb = prm[br][15];
    const float* otw = prm[br][16], *otb = prm[br][17];

    // weights -> f16, rows zero-padded to a 64-multiple so B fragments need
    // no bounds guard (tiny, L2-resident afterwards)
    cvtpad_kernel<<<48, 256, 0, stream>>>(qw, wq16, 9216, 12288);
    cvtpad_kernel<<<48, 256, 0, stream>>>(kw, wk16, 9216, 12288);
    cvtpad_kernel<<<48, 256, 0, stream>>>(vw, wv16, 9216, 12288);
    cvtpad_kernel<<<48, 256, 0, stream>>>(pw, wp16, 9216, 12288);
    cvtpad_kernel<<<72, 256, 0, stream>>>(inw, win16, 18432, 18432);
    cvtpad_kernel<<<96, 256, 0, stream>>>(otw, wot16, 18432, 24576);

    slice_mi_kernel<<<24, 256, 0, stream>>>(rm, mi16, br * 96);
    winpart_kernel<<<(M * 96) / 256, 256, 0, stream>>>(dsx, xw16, dh, dw, nWw, nWh);

    launch_gemm(stream, xw16, wq16, qb, M, 96, 96, 1.0f, 0, q16, nullptr,
                nullptr, 0, 0, 0, 0);
    launch_gemm(stream, mi16, wk16, kb, 64, 96, 96, 1.0f, 0, k16, nullptr,
                nullptr, 0, 0, 0, 0);
    launch_gemm(stream, mi16, wv16, vb, 64, 96, 96, 1.0f, 1, nullptr, v32,
                nullptr, 0, 0, 0, 0);
    lepe3_kernel<<<24, 256, 0, stream>>>(v32, lpw, lpb, lepe32);
    vtrans_kernel<<<24, 256, 0, stream>>>(v32, vT16);

    attn_win_kernel<<<M / 64, 128, 0, stream>>>(q16, k16, vT16, lepe32, o16buf,
                                                rsqrtf(96.0f));

    launch_gemm(stream, o16buf, wp16, pb, M, 96, 96, 1.0f, 1, nullptr, proj32,
                nullptr, 0, 0, 0, 0);
    ln_kernel<<<M / 8, 256, 0, stream>>>(proj32, lnw, lnb, ln16);

    launch_gemm(stream, ln16, win16, inb, M, 192, 96, 1.0f, 2, h1_16, nullptr,
                hcat16, 0, 0, 0, 0);
    dwconv5_kernel<<<(M * 96) / 256, 256, 0, stream>>>(h1_16, dcw, dcb, hcat16,
                                                       dh, dw, nWw, nWh);
    launch_gemm(stream, hcat16, wot16, otb, M, 96, 192, 1.0f, 3, nullptr, outb,
                nullptr, dh, dw, nWw, nWh);
  }
}
